// ScaledDotProductAttention_22445499089453
// MI455X (gfx1250) — matrix operand order repaired and checked by reading, not compiled, not run on hardware
//
#include <hip/hip_runtime.h>
#include <hip/hip_bf16.h>
#include <stddef.h>

// ---------------------------------------------------------------------------
// CDNA5 (gfx1250, wave32) multi-head attention:
//   QKV = x@W + b (bf16 WMMA, f32 accum), flash attention with
//   v_wmma_f32_16x16x32_bf16 for both Q@K^T and P@V.
// ---------------------------------------------------------------------------

typedef __bf16 bf16_t;
typedef bf16_t v16bf __attribute__((ext_vector_type(16)));
typedef float  v8f   __attribute__((ext_vector_type(8)));

union FragAB { unsigned int u[8]; v16bf bf; };  // 16x32 bf16 A or 32x16 bf16 B operand
union FragC  { float f[8];        v8f  v;  };   // 16x16 f32 C/D operand

static __device__ __forceinline__ unsigned short f2bf(float x) {
    unsigned int u = __float_as_uint(x);
    u += 0x7FFFu + ((u >> 16) & 1u);  // round-to-nearest-even
    return (unsigned short)(u >> 16);
}

#define EMBED 1024
#define SEQ   2048
#define BATCH 4
#define HEADS 16
#define HDIM  64
#define MROWS (BATCH * SEQ)  // 8192

// ---------------------------------------------------------------------------
// Kernel 1: x (f32) -> xbf (bf16), elementwise
// ---------------------------------------------------------------------------
__global__ void cvt_x_kernel(const float* __restrict__ x,
                             unsigned short* __restrict__ xbf, int n) {
    int i = blockIdx.x * blockDim.x + threadIdx.x;
    if (i < n) xbf[i] = f2bf(x[i]);
}

// ---------------------------------------------------------------------------
// Kernel 2: Wt[w][n][k] = bf16(W_w[k][n])   (transpose so B-operand loads are
// contiguous over the contraction index k)
// ---------------------------------------------------------------------------
__global__ void cvt_w_kernel(const float* __restrict__ Wq,
                             const float* __restrict__ Wk,
                             const float* __restrict__ Wv,
                             unsigned short* __restrict__ Wt) {
    int k = blockIdx.x * blockDim.x + threadIdx.x;   // 0..1023
    int n = blockIdx.y;                              // 0..1023
    int w = blockIdx.z;                              // 0..2
    const float* W = (w == 0) ? Wq : (w == 1) ? Wk : Wv;
    Wt[(size_t)w * EMBED * EMBED + (size_t)n * EMBED + k] = f2bf(W[(size_t)k * EMBED + n]);
}

// ---------------------------------------------------------------------------
// Kernel 3: QKV projection GEMM. One wave = one 16x16 output tile,
// K-loop in steps of 32 with v_wmma_f32_16x16x32_bf16.
//   z==0 -> Q (B,H,S,D) bf16, pre-scaled by 1/sqrt(64)
//   z==1 -> K (B,H,S,D) bf16
//   z==2 -> V (B,H,D,S) bf16 (transposed for P@V B-operand)
// ---------------------------------------------------------------------------
__global__ __launch_bounds__(128) void qkv_gemm_kernel(
    const unsigned short* __restrict__ xbf,
    const unsigned short* __restrict__ Wt,
    const float* __restrict__ bq, const float* __restrict__ bk,
    const float* __restrict__ bv,
    unsigned short* __restrict__ Qb, unsigned short* __restrict__ Kb,
    unsigned short* __restrict__ Vt) {
    const int lane = threadIdx.x & 31;
    const int wave = threadIdx.x >> 5;
    const int z    = blockIdx.z;
    const int m0   = blockIdx.y * 16;
    const int n0   = blockIdx.x * 64 + wave * 16;
    const int lm   = lane & 15;
    const int hi   = lane >> 4;     // which 16-lane half
    const int kh   = hi * 8;        // unused; A's upper-half K offset is hi * 16

    const unsigned short* Wz = Wt + (size_t)z * EMBED * EMBED;
    const float* bias = (z == 0) ? bq : (z == 1) ? bk : bv;

    const unsigned short* arow_p = xbf + (size_t)(m0 + lm) * EMBED;
    const unsigned short* bcol_p = Wz + (size_t)(n0 + lm) * EMBED;

    FragC acc;
#pragma unroll
    for (int r = 0; r < 8; ++r) acc.f[r] = 0.0f;

    for (int k0 = 0; k0 < EMBED; k0 += 32) {
        __builtin_prefetch(arow_p + k0 + 128, 0, 0);
        __builtin_prefetch(bcol_p + k0 + 128, 0, 0);
        FragAB a, b;
#pragma unroll
        for (int v = 0; v < 8; ++v) {
            // A (16x32): lanes 0-15 rows m0+lm, lanes 0-15 K=0..15, lanes 16-31 K=16..31
            int ka = k0 + hi * 16 + 2 * v;
            a.u[v] = *(const unsigned int*)(arow_p + ka);
            // B (32x16): lanes 0-15 K=0..15, lanes 16-31 K=16..31
            int kb = k0 + hi * 16 + 2 * v;
            b.u[v] = *(const unsigned int*)(bcol_p + kb);
        }
        acc.v = __builtin_amdgcn_wmma_f32_16x16x32_bf16(
            false, a.bf, false, b.bf, (short)0, acc.v, false, false);
    }

    const float bval = bias[n0 + lm];
    const int   n    = n0 + lm;
    const int   h    = n >> 6;
    const int   d    = n & 63;
#pragma unroll
    for (int r = 0; r < 8; ++r) {
        int row  = m0 + r + hi * 8;          // global row in [0, 8192)
        int bidx = row >> 11;                // batch
        int s    = row & (SEQ - 1);          // sequence pos
        float val = acc.f[r] + bval;
        size_t bh = (size_t)(bidx * HEADS + h);
        if (z == 0) {
            Qb[(bh * SEQ + s) * HDIM + d] = f2bf(val * 0.125f);  // 1/sqrt(64)
        } else if (z == 1) {
            Kb[(bh * SEQ + s) * HDIM + d] = f2bf(val);
        } else {
            Vt[(bh * HDIM + d) * SEQ + s] = f2bf(val);
        }
    }
}

// ---------------------------------------------------------------------------
// Kernel 4: flash attention. One wave = 16 query rows x full key stream.
// Per 32-key block: 4 WMMA (scores) + online softmax + LDS relayout of P
// (C-frag -> A-frag) + 4 WMMA (P @ V^T).
// ---------------------------------------------------------------------------
__global__ __launch_bounds__(128) void attn_kernel(
    const unsigned short* __restrict__ Qb,
    const unsigned short* __restrict__ Kb,
    const unsigned short* __restrict__ Vt,
    float* __restrict__ out) {
    __shared__ unsigned short Plds[4][16 * 32];  // per-wave 16x32 bf16 P tile

    const int lane  = threadIdx.x & 31;
    const int wave  = threadIdx.x >> 5;
    const int lm    = lane & 15;
    const int hi    = lane >> 4;
    const int qtile = blockIdx.x * 4 + wave;     // 0..127
    const int bh    = blockIdx.y;                // 0..63
    const int b     = bh >> 4;
    const int h     = bh & 15;
    const int q0    = qtile * 16;

    const unsigned short* Qp = Qb + (size_t)bh * SEQ * HDIM;
    const unsigned short* Kp = Kb + (size_t)bh * SEQ * HDIM;
    const unsigned short* Vp = Vt + (size_t)bh * HDIM * SEQ;
    unsigned short* Pl = Plds[wave];

    // Q fragments (A operand), 16 rows x 64 d, split into two 16x32 frags
    FragAB qa[2];
#pragma unroll
    for (int c = 0; c < 2; ++c)
#pragma unroll
        for (int v = 0; v < 8; ++v) {
            int d = 32 * c + hi * 16 + 2 * v;
            qa[c].u[v] = *(const unsigned int*)(Qp + (size_t)(q0 + lm) * HDIM + d);
        }

    FragC O[4];
#pragma unroll
    for (int f = 0; f < 4; ++f)
#pragma unroll
        for (int r = 0; r < 8; ++r) O[f].f[r] = 0.0f;

    float mi[8], li[8];
#pragma unroll
    for (int r = 0; r < 8; ++r) { mi[r] = -3.0e38f; li[r] = 0.0f; }

    for (int k0 = 0; k0 < SEQ; k0 += 32) {
        // ---- scores: two 16(q) x 16(key) tiles, contraction over d=64 ----
        FragC s0, s1;
#pragma unroll
        for (int r = 0; r < 8; ++r) { s0.f[r] = 0.0f; s1.f[r] = 0.0f; }
#pragma unroll
        for (int c = 0; c < 2; ++c) {
            FragAB kb0, kb1;
#pragma unroll
            for (int v = 0; v < 8; ++v) {
                int d = 32 * c + hi * 16 + 2 * v;   // B-operand contraction idx
                kb0.u[v] = *(const unsigned int*)(Kp + (size_t)(k0 + lm) * HDIM + d);
                kb1.u[v] = *(const unsigned int*)(Kp + (size_t)(k0 + 16 + lm) * HDIM + d);
            }
            s0.v = __builtin_amdgcn_wmma_f32_16x16x32_bf16(
                false, qa[c].bf, false, kb0.bf, (short)0, s0.v, false, false);
            s1.v = __builtin_amdgcn_wmma_f32_16x16x32_bf16(
                false, qa[c].bf, false, kb1.bf, (short)0, s1.v, false, false);
        }

        // ---- online softmax (rows live within 16-lane halves) ----
        float alpha[8];
#pragma unroll
        for (int r = 0; r < 8; ++r) {
            float sm = fmaxf(s0.f[r], s1.f[r]);
#pragma unroll
            for (int m = 1; m < 16; m <<= 1) sm = fmaxf(sm, __shfl_xor(sm, m, 32));
            float mn = fmaxf(mi[r], sm);
            float a  = __expf(mi[r] - mn);
            float p0 = __expf(s0.f[r] - mn);
            float p1 = __expf(s1.f[r] - mn);
            float rs = p0 + p1;
#pragma unroll
            for (int m = 1; m < 16; m <<= 1) rs += __shfl_xor(rs, m, 32);
            li[r]   = a * li[r] + rs;
            mi[r]   = mn;
            alpha[r] = a;
            int row = r + hi * 8;
            Pl[row * 32 + lm]      = f2bf(p0);   // keys k0..k0+15
            Pl[row * 32 + 16 + lm] = f2bf(p1);   // keys k0+16..k0+31
        }

        // per-wave LDS RAW: DS ops are in-order per wave; enforce + stop
        // compiler reordering across the relayout
        asm volatile("s_wait_dscnt 0x0" ::: "memory");

        // ---- relayout P: C-frag -> 16x32 A-frag ----
        FragAB pa;
#pragma unroll
        for (int v = 0; v < 8; ++v) {
            int k = hi * 16 + 2 * v;
            pa.u[v] = *(const unsigned int*)(Pl + lm * 32 + k);
        }

        // ---- rescale O by alpha, accumulate P @ V^T ----
#pragma unroll
        for (int f = 0; f < 4; ++f)
#pragma unroll
            for (int r = 0; r < 8; ++r) O[f].f[r] *= alpha[r];

#pragma unroll
        for (int f = 0; f < 4; ++f) {
            FragAB vb;
#pragma unroll
            for (int v = 0; v < 8; ++v) {
                int kk = hi * 16 + 2 * v;        // contraction over 32 keys
                vb.u[v] = *(const unsigned int*)(Vp + (size_t)(16 * f + lm) * SEQ + k0 + kk);
            }
            O[f].v = __builtin_amdgcn_wmma_f32_16x16x32_bf16(
                false, pa.bf, false, vb.bf, (short)0, O[f].v, false, false);
        }
    }

    // ---- normalize and store (B, S, H*D) f32 ----
#pragma unroll
    for (int f = 0; f < 4; ++f) {
#pragma unroll
        for (int r = 0; r < 8; ++r) {
            int row = q0 + r + hi * 8;
            float o = O[f].f[r] / li[r];
            out[((size_t)b * SEQ + row) * EMBED + h * HDIM + 16 * f + lm] = o;
        }
    }
}

// ---------------------------------------------------------------------------
// Host launcher
// ---------------------------------------------------------------------------
extern "C" void kernel_launch(void* const* d_in, const int* in_sizes, int n_in,
                              void* d_out, int out_size, void* d_ws, size_t ws_size,
                              hipStream_t stream) {
    const float* x  = (const float*)d_in[0];
    const float* Wq = (const float*)d_in[1];
    const float* bq = (const float*)d_in[2];
    const float* Wk = (const float*)d_in[3];
    const float* bk = (const float*)d_in[4];
    const float* Wv = (const float*)d_in[5];
    const float* bv = (const float*)d_in[6];
    float* out = (float*)d_out;

    // workspace layout (bf16 elements)
    unsigned short* xbf = (unsigned short*)d_ws;                    // 8192*1024
    unsigned short* Wt  = xbf + (size_t)MROWS * EMBED;              // 3*1024*1024
    unsigned short* Qb  = Wt + (size_t)3 * EMBED * EMBED;           // 64*2048*64
    unsigned short* Kb  = Qb + (size_t)BATCH * HEADS * SEQ * HDIM;  // 64*2048*64
    unsigned short* Vt  = Kb + (size_t)BATCH * HEADS * SEQ * HDIM;  // 64*64*2048

    // 1) x -> bf16
    {
        int n = MROWS * EMBED;
        cvt_x_kernel<<<(n + 255) / 256, 256, 0, stream>>>(x, xbf, n);
    }
    // 2) W transpose -> bf16
    {
        dim3 grid(EMBED / 256, EMBED, 3);
        cvt_w_kernel<<<grid, 256, 0, stream>>>(Wq, Wk, Wv, Wt);
    }
    // 3) QKV GEMM: grid = (N/64, M/16, 3), 4 waves/block (16 cols each)
    {
        dim3 grid(EMBED / 64, MROWS / 16, 3);
        qkv_gemm_kernel<<<grid, 128, 0, stream>>>(xbf, Wt, bq, bk, bv, Qb, Kb, Vt);
    }
    // 4) attention: grid = (128 qtiles / 4 waves, B*H)
    {
        dim3 grid((SEQ / 16) / 4, BATCH * HEADS);
        attn_kernel<<<grid, 128, 0, stream>>>(Qb, Kb, Vt, out);
    }
}
